// BlockAll_11699490914516
// MI455X (gfx1250) — matrix operand order repaired and checked by reading, not compiled, not run on hardware
//
#include <hip/hip_runtime.h>
#include <hip/hip_bf16.h>
#include <math.h>
#include <stdint.h>

// MI455X (gfx1250): all GEMMs + attention on v_wmma_f32_16x16x32_bf16,
// GEMM staging via async global->LDS (ASYNCcnt) with double buffering.

typedef __attribute__((ext_vector_type(16))) __bf16 v16bf;
typedef __attribute__((ext_vector_type(8)))  float  v8f;

union Frag16 {
  v16bf v;
  uint4 q[2];
};

#define EPS_ATTN 1e-6f

static __device__ __forceinline__ v8f wmma_bf16(const Frag16& a, const Frag16& b, v8f c) {
  return __builtin_amdgcn_wmma_f32_16x16x32_bf16(false, a.v, false, b.v, (short)0, c, false, false);
}

// async 16B global -> LDS copy, tracked by ASYNCcnt (one instr per wave)
static __device__ __forceinline__ void async_copy_b128(unsigned lds_addr, const void* gptr) {
  asm volatile("global_load_async_to_lds_b128 %0, %1, off"
               :: "v"(lds_addr), "v"((unsigned long long)(uintptr_t)gptr)
               : "memory");
}
static __device__ __forceinline__ unsigned lds_lo32(const void* p) {
  return (unsigned)(uintptr_t)p;   // low 32 bits of flat addr == LDS offset (ISA 10.2)
}

// ---------------- weight conversion ----------------
__global__ void cvt_bf16_kernel(const float* __restrict__ w, __bf16* __restrict__ o, int n) {
  int i = blockIdx.x * blockDim.x + threadIdx.x;
  if (i < n) o[i] = (__bf16)w[i];
}

__global__ void cvt_masked_bf16_kernel(const float* __restrict__ w, const int* __restrict__ m,
                                       __bf16* __restrict__ o, int n) {
  int i = blockIdx.x * blockDim.x + threadIdx.x;
  if (i < n) o[i] = (__bf16)(w[i] * (float)m[i]);
}

// ---------------- layernorm over C=1024, f32 in -> bf16 out ----------------
__global__ __launch_bounds__(256)
void ln_bf16_kernel(const float* __restrict__ x, const float* __restrict__ g,
                    const float* __restrict__ b, __bf16* __restrict__ out) {
  const int C = 1024;
  const int row = blockIdx.x;
  const int t = threadIdx.x;
  __shared__ float ssum[256];
  __shared__ float ssq[256];
  float4 xv = ((const float4*)(x + (size_t)row * C))[t];
  ssum[t] = xv.x + xv.y + xv.z + xv.w;
  ssq[t]  = xv.x*xv.x + xv.y*xv.y + xv.z*xv.z + xv.w*xv.w;
  __syncthreads();
  for (int k = 128; k > 0; k >>= 1) {
    if (t < k) { ssum[t] += ssum[t+k]; ssq[t] += ssq[t+k]; }
    __syncthreads();
  }
  float mean = ssum[0] * (1.0f / C);
  float var  = ssq[0] * (1.0f / C) - mean * mean;
  float rstd = rsqrtf(var + 1e-5f);
  float4 gv = ((const float4*)g)[t];
  float4 bv = ((const float4*)b)[t];
  __bf16* orow = out + (size_t)row * C + t * 4;
  orow[0] = (__bf16)((xv.x - mean) * rstd * gv.x + bv.x);
  orow[1] = (__bf16)((xv.y - mean) * rstd * gv.y + bv.y);
  orow[2] = (__bf16)((xv.z - mean) * rstd * gv.z + bv.z);
  orow[3] = (__bf16)((xv.w - mean) * rstd * gv.w + bv.w);
}

// ---------------- column sums of V (for the EPS/N numerator term) ----------------
__global__ __launch_bounds__(256)
void vcolsum_kernel(const __bf16* __restrict__ Vt, float* __restrict__ sumv) {
  int i = blockIdx.x * blockDim.x + threadIdx.x;   // 0..8191  (= bh*64 + d)
  const __bf16* row = Vt + (size_t)i * 1024;
  float s = 0.0f;
  for (int n = 0; n < 1024; n += 8) {
    union { uint4 u; __bf16 e[8]; } cv;
    cv.u = *(const uint4*)(row + n);
#pragma unroll
    for (int k = 0; k < 8; k++) s += (float)cv.e[k];
  }
  sumv[i] = s;
}

// ---------------- tiled bf16 WMMA GEMM: D = A[M,K] @ W[Nn,K]^T (+epilogue) ----------------
constexpr int MODE_QKV = 0;   // scatter to Q [BH,N,64], K [BH,N,64], Vt [BH,64,N] (bf16)
constexpr int MODE_PROJ = 1;  // outf = res + acc + bias  (f32)
constexpr int MODE_FC1 = 2;   // outb = gelu(acc + bias)  (bf16)
constexpr int MODE_FC2 = 3;   // outf = res + acc + bias  (f32)

// 128(M) x 128(N) per workgroup; 8 waves in 2(M)x4(N); 64x32 per wave; K-step 32.
// Double-buffered LDS tiles filled by async global->LDS; ~64 FLOP/byte of HBM traffic.
template<int MODE>
__global__ __launch_bounds__(256)
void gemm_wmma_kernel(const __bf16* __restrict__ A, const __bf16* __restrict__ W,
                      int M, int Nn, int K,
                      const float* __restrict__ bias, const float* __restrict__ res,
                      float* __restrict__ outf, __bf16* __restrict__ outb,
                      __bf16* __restrict__ qd, __bf16* __restrict__ kd, __bf16* __restrict__ vtd) {
  __shared__ __align__(16) __bf16 AsBuf[2][128 * 40];   // +8 bf16 row pad
  __shared__ __align__(16) __bf16 BsBuf[2][128 * 40];

  const int tid  = threadIdx.x;
  const int lane = tid & 31;
  const int wid  = tid >> 5;
  const int wm   = wid & 1;      // wave M index (0..1) -> 64 rows
  const int wn   = wid >> 1;     // wave N index (0..3) -> 32 cols
  const int hlf  = lane >> 4;
  const int lr   = lane & 15;
  const int bm   = blockIdx.y * 128;
  const int bn   = blockIdx.x * 128;

  v8f zero8 = {0.f, 0.f, 0.f, 0.f, 0.f, 0.f, 0.f, 0.f};
  v8f acc[4][2];
#pragma unroll
  for (int i = 0; i < 4; i++)
#pragma unroll
    for (int j = 0; j < 2; j++) acc[i][j] = zero8;

  // async-stage addressing: thread covers rows r and r+64 of both tiles, 8 bf16 each
  const int ar = tid >> 2;             // 0..63
  const int kc = (tid & 3) * 8;        // 0,8,16,24
  const __bf16* Ag0 = A + (size_t)(bm + ar) * K + kc;
  const __bf16* Ag1 = A + (size_t)(bm + ar + 64) * K + kc;
  const __bf16* Bg0 = W + (size_t)(bn + ar) * K + kc;
  const __bf16* Bg1 = W + (size_t)(bn + ar + 64) * K + kc;
  const unsigned ldsOff = (unsigned)(ar * 40 + kc) * 2u;
  const unsigned ldsOff1 = ldsOff + 64u * 40u * 2u;

  const int nstages = K >> 5;
  // prologue: fill stage 0
  {
    unsigned a0 = lds_lo32(&AsBuf[0][0]);
    unsigned b0 = lds_lo32(&BsBuf[0][0]);
    async_copy_b128(a0 + ldsOff,  Ag0);
    async_copy_b128(a0 + ldsOff1, Ag1);
    async_copy_b128(b0 + ldsOff,  Bg0);
    async_copy_b128(b0 + ldsOff1, Bg1);
  }

  for (int s = 0; s < nstages; s++) {
    const int cur = s & 1;
    if (s + 1 < nstages) {
      const int k1 = (s + 1) << 5;
      unsigned a0 = lds_lo32(&AsBuf[1 - cur][0]);
      unsigned b0 = lds_lo32(&BsBuf[1 - cur][0]);
      async_copy_b128(a0 + ldsOff,  Ag0 + k1);
      async_copy_b128(a0 + ldsOff1, Ag1 + k1);
      async_copy_b128(b0 + ldsOff,  Bg0 + k1);
      async_copy_b128(b0 + ldsOff1, Bg1 + k1);
      asm volatile("s_wait_asynccnt 4" ::: "memory");   // stage s landed, s+1 in flight
    } else {
      asm volatile("s_wait_asynccnt 0" ::: "memory");
    }
    __syncthreads();

    const __bf16* As = AsBuf[cur];
    const __bf16* Bs = BsBuf[cur];
    Frag16 a[4], b[2];
#pragma unroll
    for (int i = 0; i < 4; i++) {
      int r = wm * 64 + i * 16 + lr;
      a[i].q[0] = *(const uint4*)(&As[r * 40 + hlf * 8]);        // K = hlf*8..+7
      a[i].q[1] = *(const uint4*)(&As[r * 40 + 16 + hlf * 8]);   // K = 16+hlf*8..+7
    }
#pragma unroll
    for (int j = 0; j < 2; j++) {
      int c = wn * 32 + j * 16 + lr;
      b[j].q[0] = *(const uint4*)(&Bs[c * 40 + hlf * 8]);        // K = hlf*8..+7
      b[j].q[1] = *(const uint4*)(&Bs[c * 40 + 16 + hlf * 8]);   // K = 16+hlf*8..+7
    }
#pragma unroll
    for (int i = 0; i < 4; i++)
#pragma unroll
      for (int j = 0; j < 2; j++)
        acc[i][j] = wmma_bf16(a[i], b[j], acc[i][j]);
    __syncthreads();   // everyone done reading 'cur' before it is refilled
  }

  // Epilogue. f32 C/D layout: lane holds (m = hlf*8 + v, n = lr).
#pragma unroll
  for (int i = 0; i < 4; i++) {
#pragma unroll
    for (int j = 0; j < 2; j++) {
#pragma unroll
      for (int v = 0; v < 8; v++) {
        int gm = bm + wm * 64 + i * 16 + hlf * 8 + v;
        int gn = bn + wn * 32 + j * 16 + lr;
        float val = acc[i][j][v];
        if (MODE == MODE_QKV) {
          int s = gn >> 10, hh = (gn >> 6) & 15, d = gn & 63;
          int bb = gm >> 10, nn = gm & 1023;
          size_t bh = (size_t)bb * 16 + hh;
          if (s == 0)      qd[(bh * 1024 + nn) * 64 + d] = (__bf16)val;
          else if (s == 1) kd[(bh * 1024 + nn) * 64 + d] = (__bf16)val;
          else             vtd[(bh * 64 + d) * 1024 + nn] = (__bf16)val;  // V transposed
        } else if (MODE == MODE_FC1) {
          float t = val + bias[gn];
          float ge = 0.5f * t * (1.0f + erff(t * 0.70710678118654752f)); // exact GELU
          outb[(size_t)gm * Nn + gn] = (__bf16)ge;
        } else {  // MODE_PROJ / MODE_FC2
          size_t idx = (size_t)gm * Nn + gn;
          outf[idx] = res[idx] + val + bias[gn];
        }
      }
    }
  }
}

// ---------------- flash attention with policy mask, WMMA QK^T and PV ----------------
__global__ __launch_bounds__(256)
void attn_kernel(const __bf16* __restrict__ Q, const __bf16* __restrict__ Km,
                 const __bf16* __restrict__ Vt, const float* __restrict__ policy,
                 const float* __restrict__ sumv, __bf16* __restrict__ O) {
  const int N = 1024, DH = 64;
  const float scale = 0.125f;            // dh^-0.5, dh=64
  const int bh   = blockIdx.y;           // 0..127
  const int bidx = bh >> 4, h = bh & 15;
  const int w    = threadIdx.x >> 5;     // one wave = 16 query rows
  const int lane = threadIdx.x & 31;
  const int hlf  = lane >> 4, lr = lane & 15;
  const int mbase = blockIdx.x * 128 + w * 16;

  __shared__ __align__(16) __bf16 Pbuf[8][16 * 32];  // per-wave P tile

  v8f zero8 = {0.f, 0.f, 0.f, 0.f, 0.f, 0.f, 0.f, 0.f};

  Frag16 aq[2];
  const char* qrow = (const char*)(Q + ((size_t)bh * N + mbase + lr) * DH);
#pragma unroll
  for (int c = 0; c < 2; c++) {
    aq[c].q[0] = *(const uint4*)(qrow + c * 64 + hlf * 32);
    aq[c].q[1] = *(const uint4*)(qrow + c * 64 + hlf * 32 + 16);
  }

  v8f o[4];
#pragma unroll
  for (int t = 0; t < 4; t++) o[t] = zero8;
  float mrun[8], lrun[8];
#pragma unroll
  for (int v = 0; v < 8; v++) { mrun[v] = -3.0e38f; lrun[v] = 0.0f; }

  const float* pol = policy + (size_t)bidx * N;

  for (int j0 = 0; j0 < N; j0 += 32) {
    v8f s0 = zero8, s1 = zero8;
#pragma unroll
    for (int t = 0; t < 2; t++) {
      const char* krow = (const char*)(Km + ((size_t)bh * N + j0 + t * 16 + lr) * DH);
      Frag16 bk0, bk1;
      bk0.q[0] = *(const uint4*)(krow + hlf * 32);
      bk0.q[1] = *(const uint4*)(krow + hlf * 32 + 16);
      bk1.q[0] = *(const uint4*)(krow + 64 + hlf * 32);
      bk1.q[1] = *(const uint4*)(krow + 64 + hlf * 32 + 16);
      v8f s = zero8;
      s = wmma_bf16(aq[0], bk0, s);
      s = wmma_bf16(aq[1], bk1, s);
      if (t == 0) s0 = s; else s1 = s;
    }

    float pA = pol[j0 + lr];
    float pB = pol[j0 + 16 + lr];
    float alpha_v[8];
#pragma unroll
    for (int v = 0; v < 8; v++) {
      int mg = mbase + hlf * 8 + v;
      float sa = s0[v] * scale, sb = s1[v] * scale;
      float cand = fmaxf(sa, sb);
      for (int off = 1; off < 16; off <<= 1) cand = fmaxf(cand, __shfl_xor(cand, off, 16));
      float mnew  = fmaxf(mrun[v], cand);
      float alpha = expf(mrun[v] - mnew);
      float apA = pA + (1.0f - pA) * ((j0 + lr)      == mg ? 1.0f : 0.0f);  // self-attend
      float apB = pB + (1.0f - pB) * ((j0 + 16 + lr) == mg ? 1.0f : 0.0f);
      float pa = expf(sa - mnew) * apA;
      float pb = expf(sb - mnew) * apB;
      float rs = pa + pb;
      for (int off = 1; off < 16; off <<= 1) rs += __shfl_xor(rs, off, 16);
      lrun[v] = lrun[v] * alpha + rs;
      mrun[v] = mnew;
      alpha_v[v] = alpha;
      Pbuf[w][(hlf * 8 + v) * 32 + lr]      = (__bf16)pa;
      Pbuf[w][(hlf * 8 + v) * 32 + 16 + lr] = (__bf16)pb;
    }
#pragma unroll
    for (int t = 0; t < 4; t++)
#pragma unroll
      for (int v = 0; v < 8; v++) o[t][v] *= alpha_v[v];

    Frag16 pf;   // P as 16x32 A-fragment (intra-wave LDS round trip)
    pf.q[0] = *(const uint4*)(&Pbuf[w][lr * 32 + hlf * 8]);
    pf.q[1] = *(const uint4*)(&Pbuf[w][lr * 32 + 16 + hlf * 8]);

#pragma unroll
    for (int t = 0; t < 4; t++) {
      const char* vrow = (const char*)(Vt + ((size_t)bh * DH + t * 16 + lr) * N + j0);
      Frag16 vb;
      vb.q[0] = *(const uint4*)(vrow + hlf * 16);
      vb.q[1] = *(const uint4*)(vrow + hlf * 16 + 32);
      o[t] = wmma_bf16(pf, vb, o[t]);
    }
  }

  const float eps_n = EPS_ATTN / 1024.0f;
#pragma unroll
  for (int t = 0; t < 4; t++) {
    float sv = sumv[(size_t)bh * DH + t * 16 + lr];
#pragma unroll
    for (int v = 0; v < 8; v++) {
      int nrow = mbase + hlf * 8 + v;
      float val = (o[t][v] + eps_n * sv) / (lrun[v] + EPS_ATTN);
      O[((size_t)bidx * N + nrow) * 1024 + h * 64 + t * 16 + lr] = (__bf16)val;
    }
  }
}

// ---------------- launch ----------------
extern "C" void kernel_launch(void* const* d_in, const int* in_sizes, int n_in,
                              void* d_out, int out_size, void* d_ws, size_t ws_size,
                              hipStream_t stream) {
  (void)in_sizes; (void)n_in; (void)out_size; (void)ws_size;
  const int B = 8, N = 1024, C = 1024, F = 4096;
  const size_t BN = (size_t)B * N;  // 8192

  const float* x      = (const float*)d_in[0];
  const float* policy = (const float*)d_in[1];
  const float* ln1_g  = (const float*)d_in[2];
  const float* ln1_b  = (const float*)d_in[3];
  const float* qkv_w  = (const float*)d_in[4];
  const float* proj_w = (const float*)d_in[5];
  const float* proj_b = (const float*)d_in[6];
  const float* ln2_g  = (const float*)d_in[7];
  const float* ln2_b  = (const float*)d_in[8];
  const float* fc1_w  = (const float*)d_in[9];
  const float* fc1_b  = (const float*)d_in[10];
  const float* fc2_w  = (const float*)d_in[11];
  const float* fc2_b  = (const float*)d_in[12];
  const int*   mask1  = (const int*)d_in[13];
  const int*   mask2  = (const int*)d_in[14];
  float* out = (float*)d_out;

  char* ws = (char*)d_ws;
  size_t off = 0;
  auto take = [&](size_t bytes) -> char* {
    char* p = ws + off;
    off += (bytes + 255) & ~(size_t)255;
    return p;
  };
  __bf16* wqkvb = (__bf16*)take((size_t)3 * C * C * 2);
  __bf16* wprojb = (__bf16*)take((size_t)C * C * 2);
  __bf16* w1b   = (__bf16*)take((size_t)F * C * 2);
  __bf16* w2b   = (__bf16*)take((size_t)C * F * 2);
  float*  x1    = (float*)take(BN * C * 4);
  float*  sumvb = (float*)take((size_t)B * 16 * 64 * 4);
  __bf16* hb    = (__bf16*)take(BN * C * 2);   // LN1 out; reused as attention output
  __bf16* Qb    = (__bf16*)take(BN * C * 2);   // reused as LN2 out
  __bf16* Kb    = (__bf16*)take(BN * C * 2);
  __bf16* Vtb   = (__bf16*)take(BN * C * 2);
  __bf16* yb    = (__bf16*)take(BN * F * 2);
  __bf16* ob    = hb;
  __bf16* h2b   = Qb;

  const int T = 256;
  cvt_bf16_kernel<<<(3 * C * C) / T, T, 0, stream>>>(qkv_w, wqkvb, 3 * C * C);
  cvt_bf16_kernel<<<(C * C) / T, T, 0, stream>>>(proj_w, wprojb, C * C);
  cvt_masked_bf16_kernel<<<(F * C) / T, T, 0, stream>>>(fc1_w, mask1, w1b, F * C);
  cvt_masked_bf16_kernel<<<(C * F) / T, T, 0, stream>>>(fc2_w, mask2, w2b, C * F);

  ln_bf16_kernel<<<(int)BN, 256, 0, stream>>>(x, ln1_g, ln1_b, hb);

  gemm_wmma_kernel<MODE_QKV><<<dim3(3072 / 128, 8192 / 128), 256, 0, stream>>>(
      hb, wqkvb, 8192, 3072, 1024, nullptr, nullptr, nullptr, nullptr, Qb, Kb, Vtb);

  vcolsum_kernel<<<8192 / 256, 256, 0, stream>>>(Vtb, sumvb);

  attn_kernel<<<dim3(1024 / 128, B * 16), 256, 0, stream>>>(Qb, Kb, Vtb, policy, sumvb, ob);

  gemm_wmma_kernel<MODE_PROJ><<<dim3(1024 / 128, 8192 / 128), 256, 0, stream>>>(
      ob, wprojb, 8192, 1024, 1024, proj_b, x, x1, nullptr, nullptr, nullptr, nullptr);

  ln_bf16_kernel<<<(int)BN, 256, 0, stream>>>(x1, ln2_g, ln2_b, h2b);

  gemm_wmma_kernel<MODE_FC1><<<dim3(4096 / 128, 8192 / 128), 256, 0, stream>>>(
      h2b, w1b, 8192, 4096, 1024, fc1_b, nullptr, nullptr, yb, nullptr, nullptr, nullptr);

  gemm_wmma_kernel<MODE_FC2><<<dim3(1024 / 128, 8192 / 128), 256, 0, stream>>>(
      yb, w2b, 8192, 1024, 4096, fc2_b, x1, out, nullptr, nullptr, nullptr, nullptr);
}
